// SemanticTransformer_11871289606890
// MI455X (gfx1250) — matrix operand order repaired and checked by reading, not compiled, not run on hardware
//
#include <hip/hip_runtime.h>
#include <hip/hip_bf16.h>
#include <cstdint>

// ---------------------------------------------------------------------------
// Types for CDNA5 WMMA (gfx1250, wave32)
// ---------------------------------------------------------------------------
typedef __attribute__((ext_vector_type(16))) __bf16 v16bf;
typedef __attribute__((ext_vector_type(8)))  __bf16 v8bf;
typedef __attribute__((ext_vector_type(4)))  __bf16 v4bf;
typedef __attribute__((ext_vector_type(8)))  float  v8f;

union AF {            // one 16x32 bf16 A/B fragment (8 VGPRs / lane)
    v16bf  v;
    v8bf   h[2];
    __bf16 e[16];
};

__device__ __forceinline__ v8f zero8() {
    v8f z;
#pragma unroll
    for (int i = 0; i < 8; ++i) z[i] = 0.0f;
    return z;
}

__device__ __forceinline__ v8f wmma_bf16(v16bf a, v16bf b, v8f c) {
    return __builtin_amdgcn_wmma_f32_16x16x32_bf16(
        false, a, false, b, (short)0, c, false, false);
}

__device__ __forceinline__ unsigned ldsoff(const void* p) {
    return (unsigned)(uintptr_t)p;   // low 32 bits of generic addr = LDS offset
}

// CDNA5 async copy: 16B global -> LDS, ASYNCcnt-tracked (ISA 15.18.3 op 98).
// INST_OFFSET is added to BOTH the LDS and global addresses, so callers copy
// row segments where both sides advance together.
#define ASYNC_CP16(ldsaddr, gptr, OFFSTR)                                   \
    asm volatile("global_load_async_to_lds_b128 %0, %1, off offset:" OFFSTR \
                 :: "v"(ldsaddr), "v"(gptr) : "memory")

__device__ __forceinline__ void async_wait0()   { asm volatile("s_wait_asynccnt 0" ::: "memory"); }
__device__ __forceinline__ void async_wait_le6(){ asm volatile("s_wait_asynccnt 6" ::: "memory"); }
__device__ __forceinline__ void async_wait_le2(){ asm volatile("s_wait_asynccnt 2" ::: "memory"); }

// ---------------------------------------------------------------------------
// GEMM: C[R x Ncols] = act( A[R x K] * W[K x Ncols] )
//   A  : bf16, row-major (R x K)
//   Wt : bf16, pre-transposed row-major (Ncols x K)
//   Cf / Cbf : optional fp32 / bf16 outputs
// Block = 128 threads (4 waves); tile 128x64; wave computes 32x64
// -> 8 WMMAs per 32-wide K-step per wave. Double-buffered async staging:
// 6 async b128 per wave per tile; wait asynccnt<=6 overlaps next-tile loads.
// ---------------------------------------------------------------------------
__device__ __forceinline__ void gemm_stage(
    const __bf16* __restrict__ A, const __bf16* __restrict__ Wt,
    long rowbase, long colbase, int Kdim, int k0, int tid,
    __bf16 (*As)[32], __bf16 (*Ws)[32])
{
    // A tile: thread 'tid' copies its full 64B row (4 packets, shared offsets)
    const __bf16* ga = &A[(rowbase + tid) * (long)Kdim + k0];
    unsigned la = ldsoff(&As[tid][0]);
    ASYNC_CP16(la, ga, "0");
    ASYNC_CP16(la, ga, "16");
    ASYNC_CP16(la, ga, "32");
    ASYNC_CP16(la, ga, "48");
    // W tile: 64 rows, 2 threads per row (half-row = 2 packets)
    const __bf16* gw = &Wt[(colbase + (tid >> 1)) * (long)Kdim + k0 + (tid & 1) * 16];
    unsigned lw = ldsoff(&Ws[tid >> 1][(tid & 1) * 16]);
    ASYNC_CP16(lw, gw, "0");
    ASYNC_CP16(lw, gw, "16");
}

__global__ __launch_bounds__(128) void gemm_bf16_wmma(
    const __bf16* __restrict__ A, const __bf16* __restrict__ Wt,
    float* __restrict__ Cf, __bf16* __restrict__ Cbf,
    int Kdim, int Ncols, int relu)
{
    __shared__ __align__(16) __bf16 As[2][128][32];   // 2 x 8KB
    __shared__ __align__(16) __bf16 Ws[2][64][32];    // 2 x 4KB
    const int tid  = threadIdx.x;
    const int wave = tid >> 5;
    const int lane = tid & 31;
    const int lh   = lane & 15;
    const bool hiL = lane >= 16;
    const long rowbase = (long)blockIdx.y * 128;
    const long colbase = (long)blockIdx.x * 64;
    const int mrow = wave * 32;

    v8f acc[8];
#pragma unroll
    for (int i = 0; i < 8; ++i) acc[i] = zero8();

    // A k order used here (same as B): lanes0-15: K = i (0..15)
    //                              lanes16-31: K = i+16 (16..31)
    const int aLo = hiL ? 16 : 0;
    const int aHi = hiL ? 24 : 8;
    // B k order used here (contiguous): lanes0-15 hold K=0..15, lanes16-31 hold K=16..31
    const int bBase = hiL ? 16 : 0;

    const int nsteps = Kdim >> 5;
    gemm_stage(A, Wt, rowbase, colbase, Kdim, 0, tid, As[0], Ws[0]);

    for (int s = 0; s < nsteps; ++s) {
        const int cur = s & 1;
        const bool pf = (s + 1 < nsteps);
        if (pf) {
            gemm_stage(A, Wt, rowbase, colbase, Kdim, (s + 1) * 32, tid,
                       As[cur ^ 1], Ws[cur ^ 1]);
            async_wait_le6();     // in-order completion => current tile ready
        } else {
            async_wait0();
        }
        __syncthreads();

        AF fa0, fa1;
        fa0.h[0] = *(const v8bf*)&As[cur][mrow + lh][aLo];
        fa0.h[1] = *(const v8bf*)&As[cur][mrow + lh][aHi];
        fa1.h[0] = *(const v8bf*)&As[cur][mrow + 16 + lh][aLo];
        fa1.h[1] = *(const v8bf*)&As[cur][mrow + 16 + lh][aHi];
#pragma unroll
        for (int nt = 0; nt < 4; ++nt) {
            AF fb;
            fb.h[0] = *(const v8bf*)&Ws[cur][nt * 16 + lh][bBase];
            fb.h[1] = *(const v8bf*)&Ws[cur][nt * 16 + lh][bBase + 8];
            acc[nt]     = wmma_bf16(fa0.v, fb.v, acc[nt]);
            acc[4 + nt] = wmma_bf16(fa1.v, fb.v, acc[4 + nt]);
        }
        __syncthreads();
    }

    // C layout: VGPR r -> row r (lanes 0-15) / r+8 (lanes 16-31), col = lh
#pragma unroll
    for (int g = 0; g < 2; ++g)
#pragma unroll
    for (int nt = 0; nt < 4; ++nt)
#pragma unroll
    for (int r = 0; r < 8; ++r) {
        long row = rowbase + mrow + g * 16 + r + (hiL ? 8 : 0);
        long col = colbase + nt * 16 + lh;
        float vv = acc[g * 4 + nt][r];
        if (relu) vv = fmaxf(vv, 0.0f);
        if (Cf)  Cf [row * (long)Ncols + col] = vv;
        if (Cbf) Cbf[row * (long)Ncols + col] = (__bf16)vv;
    }
}

// ---------------------------------------------------------------------------
// Flash attention: Q,K,V bf16 (B, L|S, H=4, D=64); O bf16 same layout.
// Grid (L/64, H, B); block 128 (4 waves x 16 query rows).
// Double-buffered K/V tiles, online softmax over 32-key blocks.
// Masks are identity in this configuration.
// ---------------------------------------------------------------------------
__device__ __forceinline__ void flash_stage(
    const __bf16* __restrict__ K, const __bf16* __restrict__ V,
    long baseoff /* ((b*S+s0)*H+hh)*D */, int tid,
    __bf16 (*Ks)[64], __bf16 (*Vt)[32])
{
    // K tile: 32 rows x 128B; 4 threads per row; 2 packets each
    {
        int kk = tid >> 2, seg = (tid & 3) * 16;
        const __bf16* gk = &K[baseoff + (long)kk * 256 + seg];
        unsigned lk = ldsoff(&Ks[kk][seg]);
        ASYNC_CP16(lk, gk, "0");
        ASYNC_CP16(lk, gk, "16");
    }
    // V tile transposed: vector load 8, scatter-store b16
#pragma unroll
    for (int it = 0; it < 2; ++it) {
        int t = tid + it * 128;
        int vk = t >> 3, d8 = (t & 7) * 8;
        v8bf vv = *(const v8bf*)&V[baseoff + (long)vk * 256 + d8];
#pragma unroll
        for (int i = 0; i < 8; ++i) Vt[d8 + i][vk] = vv[i];
    }
}

__global__ __launch_bounds__(128) void flash_attn_k(
    const __bf16* __restrict__ Q, const __bf16* __restrict__ K,
    const __bf16* __restrict__ V, __bf16* __restrict__ O, int L, int S)
{
    const int H = 4, D = 64;
    __shared__ __align__(16) __bf16 Ks[2][32][64];
    __shared__ __align__(16) __bf16 Vt[2][64][32];
    __shared__ __align__(16) __bf16 Ps[4][16][32];

    const int tid  = threadIdx.x;
    const int wave = tid >> 5;
    const int lane = tid & 31;
    const int lh   = lane & 15;
    const bool hiL = lane >= 16;
    const int b  = blockIdx.z;
    const int hh = blockIdx.y;

    const int dLo = hiL ? 16 : 0, dHi = hiL ? 24 : 8;
    const int bBase = hiL ? 16 : 0;

    const int qrow = blockIdx.x * 64 + wave * 16 + lh;
    const __bf16* qp = Q + ((long)(b * L + qrow) * H + hh) * D;
    AF qa0, qa1;
    qa0.h[0] = *(const v8bf*)&qp[dLo];
    qa0.h[1] = *(const v8bf*)&qp[dHi];
    qa1.h[0] = *(const v8bf*)&qp[32 + dLo];
    qa1.h[1] = *(const v8bf*)&qp[32 + dHi];

    float m[8], ssum[8];
    v8f oacc[4];
#pragma unroll
    for (int r = 0; r < 8; ++r) { m[r] = -3.0e38f; ssum[r] = 0.0f; }
#pragma unroll
    for (int i = 0; i < 4; ++i) oacc[i] = zero8();

    const long hbase = ((long)b * S * H + hh) * D;
    const int nsteps = S >> 5;
    flash_stage(K, V, hbase, tid, Ks[0], Vt[0]);

    for (int s = 0; s < nsteps; ++s) {
        const int cur = s & 1;
        const bool pf = (s + 1 < nsteps);
        if (pf) {
            flash_stage(K, V, hbase + (long)(s + 1) * 32 * H * D, tid,
                        Ks[cur ^ 1], Vt[cur ^ 1]);
            async_wait_le2();
        } else {
            async_wait0();
        }
        __syncthreads();

        // S = Q K^T : 16 rows x 32 keys via 4 WMMAs
        v8f sc0 = zero8(), sc1 = zero8();
        {
            AF kb;
            kb.h[0] = *(const v8bf*)&Ks[cur][lh][bBase];
            kb.h[1] = *(const v8bf*)&Ks[cur][lh][bBase + 8];
            sc0 = wmma_bf16(qa0.v, kb.v, sc0);
            kb.h[0] = *(const v8bf*)&Ks[cur][lh][32 + bBase];
            kb.h[1] = *(const v8bf*)&Ks[cur][lh][32 + bBase + 8];
            sc0 = wmma_bf16(qa1.v, kb.v, sc0);
            kb.h[0] = *(const v8bf*)&Ks[cur][16 + lh][bBase];
            kb.h[1] = *(const v8bf*)&Ks[cur][16 + lh][bBase + 8];
            sc1 = wmma_bf16(qa0.v, kb.v, sc1);
            kb.h[0] = *(const v8bf*)&Ks[cur][16 + lh][32 + bBase];
            kb.h[1] = *(const v8bf*)&Ks[cur][16 + lh][32 + bBase + 8];
            sc1 = wmma_bf16(qa1.v, kb.v, sc1);
        }

        // Online softmax; row reductions across 16-lane halves
#pragma unroll
        for (int r = 0; r < 8; ++r) {
            float x0 = sc0[r] * 0.125f;
            float x1 = sc1[r] * 0.125f;
            float mx = fmaxf(x0, x1);
            mx = fmaxf(mx, __shfl_xor(mx, 1, 32));
            mx = fmaxf(mx, __shfl_xor(mx, 2, 32));
            mx = fmaxf(mx, __shfl_xor(mx, 4, 32));
            mx = fmaxf(mx, __shfl_xor(mx, 8, 32));
            float nm   = fmaxf(m[r], mx);
            float corr = __expf(m[r] - nm);
            m[r] = nm;
            float p0 = __expf(x0 - nm);
            float p1 = __expf(x1 - nm);
            float ps = p0 + p1;
            ps += __shfl_xor(ps, 1, 32);
            ps += __shfl_xor(ps, 2, 32);
            ps += __shfl_xor(ps, 4, 32);
            ps += __shfl_xor(ps, 8, 32);
            ssum[r] = ssum[r] * corr + ps;
#pragma unroll
            for (int dt = 0; dt < 4; ++dt) oacc[dt][r] *= corr;
            int rr = r + (hiL ? 8 : 0);
            Ps[wave][rr][lh]      = (__bf16)p0;
            Ps[wave][rr][16 + lh] = (__bf16)p1;
        }
        // per-wave LDS RAW (C-layout -> A-layout); LDS in-order per wave
        asm volatile("s_wait_dscnt 0" ::: "memory");

        AF pf2;
        pf2.h[0] = *(const v8bf*)&Ps[wave][lh][dLo];
        pf2.h[1] = *(const v8bf*)&Ps[wave][lh][dHi];
#pragma unroll
        for (int dt = 0; dt < 4; ++dt) {
            AF vb;
            vb.h[0] = *(const v8bf*)&Vt[cur][dt * 16 + lh][bBase];
            vb.h[1] = *(const v8bf*)&Vt[cur][dt * 16 + lh][bBase + 8];
            oacc[dt] = wmma_bf16(pf2.v, vb.v, oacc[dt]);
        }
        __syncthreads();
    }

#pragma unroll
    for (int r = 0; r < 8; ++r) {
        float inv = 1.0f / ssum[r];
        int row = blockIdx.x * 64 + wave * 16 + r + (hiL ? 8 : 0);
        __bf16* op = O + ((long)(b * L + row) * H + hh) * D;
#pragma unroll
        for (int dt = 0; dt < 4; ++dt) op[dt * 16 + lh] = (__bf16)(oacc[dt][r] * inv);
    }
}

// ---------------------------------------------------------------------------
// LayerNorm over last dim C (=256): Y = (Res ? Res + ln(X) : ln(X)), fp32.
// ---------------------------------------------------------------------------
__global__ __launch_bounds__(256) void layernorm_k(
    const float* __restrict__ X, const float* __restrict__ Res,
    float* __restrict__ Y, int C)
{
    int wave = threadIdx.x >> 5, lane = threadIdx.x & 31;
    long row = (long)blockIdx.x * 8 + wave;
    const float* x = X + row * C;
    float s = 0.0f, s2 = 0.0f;
    for (int c = lane; c < C; c += 32) { float vv = x[c]; s += vv; s2 += vv * vv; }
    for (int off = 16; off >= 1; off >>= 1) {
        s  += __shfl_xor(s,  off, 32);
        s2 += __shfl_xor(s2, off, 32);
    }
    float mean = s / C;
    float var  = s2 / C - mean * mean;
    float rstd = rsqrtf(var + 1e-5f);
    float* y = Y + row * C;
    for (int c = lane; c < C; c += 32) {
        float vv = (x[c] - mean) * rstd;
        if (Res) vv += Res[row * C + c];
        y[c] = vv;
    }
}

__global__ void add_k(float* __restrict__ x, const float* __restrict__ a, long n) {
    long i = (long)blockIdx.x * 256 + threadIdx.x;
    if (i < n) x[i] += a[i];
}

// fp32 -> bf16, 4 elements / thread (vectorized)
__global__ void cvt_bf16_k(const float* __restrict__ x, __bf16* __restrict__ y, long n4) {
    long i = (long)blockIdx.x * 256 + threadIdx.x;
    if (i >= n4) return;
    float4 v = ((const float4*)x)[i];
    v4bf o;
    o[0] = (__bf16)v.x; o[1] = (__bf16)v.y; o[2] = (__bf16)v.z; o[3] = (__bf16)v.w;
    *(v4bf*)&y[i * 4] = o;
}

// Wt[n*K + k] = (bf16) W[k*N + n]   (one-time weight transpose+convert)
__global__ void transpose_cvt_k(const float* __restrict__ W, __bf16* __restrict__ Wt,
                                int K, int N) {
    long i = (long)blockIdx.x * 256 + threadIdx.x;
    if (i >= (long)K * N) return;
    long n = i / K, k = i % K;
    Wt[i] = (__bf16)W[k * (long)N + n];
}

// h_bf[r][0:256] = x[r]; h_bf[r][256:512] = msg[r]
__global__ void concat_bf_k(const float* __restrict__ x, const float* __restrict__ msg,
                            __bf16* __restrict__ h, long total) {
    long i = (long)blockIdx.x * 256 + threadIdx.x;
    if (i >= total) return;
    long row = i >> 9;
    int  cc  = (int)(i & 511);
    float vv = (cc < 256) ? x[row * 256 + cc] : msg[row * 256 + (cc - 256)];
    h[i] = (__bf16)vv;
}

// out[g][l][c] = idx<N ? feat[g/maskNum][idx][c] : 0   (pad index N -> zeros)
__global__ __launch_bounds__(256) void gather_k(
    const float* __restrict__ feat, const int* __restrict__ idx,
    float* __restrict__ out, int N, int Lp, int maskNum)
{
    long row = blockIdx.x;              // g*Lp + l
    int g = (int)(row / Lp);
    int b = g / maskNum;
    int id = idx[row];
    int c = threadIdx.x;
    float vv = 0.0f;
    if (id < N) vv = feat[((long)b * N + id) * 256 + c];
    out[row * 256 + c] = vv;
}

// ---------------------------------------------------------------------------
// Host-side orchestration
// ---------------------------------------------------------------------------
namespace {

struct Ctx {
    hipStream_t stream;
    __bf16 *xA, *sA, *qb, *kb, *vb, *ob, *hb, *t1b;
    float *ta, *tb;
};

// xout = x + ln( MLP([x, ln(attn(x,src) @ mW)]) )
void run_attn(const Ctx& c, const float* x, const float* src, float* xout,
              int Bn, int L, int S,
              const __bf16* wq, const __bf16* wk, const __bf16* wv,
              const __bf16* wm, const __bf16* w1t, const __bf16* w2t)
{
    const int C = 256;
    const long R  = (long)Bn * L;
    const long Rs = (long)Bn * S;
    hipStream_t st = c.stream;
    dim3 blk(128);

    cvt_bf16_k<<<dim3((unsigned)(R * C / 4 / 256)), dim3(256), 0, st>>>(x, c.xA, R * C / 4);
    const __bf16* srcA = c.xA;
    if (src != x) {
        cvt_bf16_k<<<dim3((unsigned)(Rs * C / 4 / 256)), dim3(256), 0, st>>>(src, c.sA, Rs * C / 4);
        srcA = c.sA;
    }
    gemm_bf16_wmma<<<dim3(C / 64, (unsigned)(R / 128)),  blk, 0, st>>>(c.xA, wq, nullptr, c.qb, C, C, 0);
    gemm_bf16_wmma<<<dim3(C / 64, (unsigned)(Rs / 128)), blk, 0, st>>>(srcA, wk, nullptr, c.kb, C, C, 0);
    gemm_bf16_wmma<<<dim3(C / 64, (unsigned)(Rs / 128)), blk, 0, st>>>(srcA, wv, nullptr, c.vb, C, C, 0);
    flash_attn_k<<<dim3(L / 64, 4, Bn), blk, 0, st>>>(c.qb, c.kb, c.vb, c.ob, L, S);
    gemm_bf16_wmma<<<dim3(C / 64, (unsigned)(R / 128)), blk, 0, st>>>(c.ob, wm, c.ta, nullptr, C, C, 0);
    layernorm_k<<<dim3((unsigned)(R / 8)), dim3(256), 0, st>>>(c.ta, nullptr, c.tb, C);     // msg
    concat_bf_k<<<dim3((unsigned)(R * 512 / 256)), dim3(256), 0, st>>>(x, c.tb, c.hb, R * 512);
    gemm_bf16_wmma<<<dim3(512 / 64, (unsigned)(R / 128)), blk, 0, st>>>(c.hb, w1t, nullptr, c.t1b, 512, 512, 1);
    gemm_bf16_wmma<<<dim3(C / 64, (unsigned)(R / 128)),  blk, 0, st>>>(c.t1b, w2t, c.ta, nullptr, 512, C, 0);
    layernorm_k<<<dim3((unsigned)(R / 8)), dim3(256), 0, st>>>(c.ta, x, xout, C);           // x + ln
}

} // namespace

extern "C" void kernel_launch(void* const* d_in, const int* in_sizes, int n_in,
                              void* d_out, int out_size, void* d_ws, size_t ws_size,
                              hipStream_t stream)
{
    (void)in_sizes; (void)n_in; (void)out_size; (void)ws_size;
    const int B = 2, N = 2048, C = 256, D2 = 384, G = 16, Lp = 512;
    const long RC = (long)B * N * C;        // 1,048,576
    const long GC = (long)G * Lp * C;       // 2,097,152

    const float* src_feat   = (const float*)d_in[0];
    const float* tgt_feat   = (const float*)d_in[1];
    const int*   si         = (const int*)  d_in[6];
    const int*   ti         = (const int*)  d_in[7];
    const float* src_2dfeat = (const float*)d_in[11];
    const float* tgt_2dfeat = (const float*)d_in[12];
    const float* dinoW      = (const float*)d_in[14];
    const float* qW         = (const float*)d_in[15];
    const float* kW         = (const float*)d_in[16];
    const float* vW         = (const float*)d_in[17];
    const float* mW         = (const float*)d_in[18];
    const float* w1         = (const float*)d_in[19];
    const float* w2         = (const float*)d_in[20];

    // -------- workspace carve --------
    float* f = (float*)d_ws;
    auto falloc = [&](long n) { float* p = f; f += n; return p; };
    float* xs  = falloc(RC);
    float* xt  = falloc(RC);
    float* s2  = falloc(RC);
    float* t2f = falloc(RC);
    float* sb  = falloc(GC);
    float* tsb = falloc(GC);
    Ctx c;
    c.stream = stream;
    c.ta = falloc(GC);
    c.tb = falloc(GC);
    __bf16* bf = (__bf16*)f;
    auto balloc = [&](long n) { __bf16* p = bf; bf += n; return p; };
    c.xA  = balloc(GC); c.sA  = balloc(GC);
    c.qb  = balloc(GC); c.kb  = balloc(GC); c.vb = balloc(GC); c.ob = balloc(GC);
    c.hb  = balloc(2 * GC); c.t1b = balloc(2 * GC);
    __bf16* wdino = balloc((long)D2 * C);
    __bf16* wqT[5]; __bf16* wkT[5]; __bf16* wvT[5]; __bf16* wmT[5];
    __bf16* w1T[5]; __bf16* w2T[5];
    for (int i = 0; i < 5; ++i) {
        wqT[i] = balloc((long)C * C);
        wkT[i] = balloc((long)C * C);
        wvT[i] = balloc((long)C * C);
        wmT[i] = balloc((long)C * C);
        w1T[i] = balloc(512L * 512);
        w2T[i] = balloc(512L * 256);
    }

    float* out_src = (float*)d_out;
    float* out_tgt = out_src + RC;
    float* out_ss  = out_tgt + RC;
    float* out_ts  = out_ss + GC;

    // -------- weight prep: fp32 -> bf16 transposed [n][k] --------
    auto tcv = [&](const float* W, __bf16* Wt, int K, int Nn) {
        long tot = (long)K * Nn;
        transpose_cvt_k<<<dim3((unsigned)((tot + 255) / 256)), dim3(256), 0, stream>>>(W, Wt, K, Nn);
    };
    tcv(dinoW, wdino, D2, C);
    for (int i = 0; i < 5; ++i) {
        tcv(qW + (long)i * C * C, wqT[i], C, C);
        tcv(kW + (long)i * C * C, wkT[i], C, C);
        tcv(vW + (long)i * C * C, wvT[i], C, C);
        tcv(mW + (long)i * C * C, wmT[i], C, C);
        tcv(w1 + (long)i * 512 * 512, w1T[i], 512, 512);
        tcv(w2 + (long)i * 512 * 256, w2T[i], 512, 256);
    }

    hipMemcpyAsync(xs, src_feat, RC * sizeof(float), hipMemcpyDeviceToDevice, stream);
    hipMemcpyAsync(xt, tgt_feat, RC * sizeof(float), hipMemcpyDeviceToDevice, stream);

    // -------- dino 2D-feature projection --------
    dim3 blk(128);
    cvt_bf16_k<<<dim3((unsigned)((long)B * N * D2 / 4 / 256)), dim3(256), 0, stream>>>(
        src_2dfeat, c.xA, (long)B * N * D2 / 4);
    gemm_bf16_wmma<<<dim3(C / 64, (unsigned)(B * N / 128)), blk, 0, stream>>>(c.xA, wdino, s2, nullptr, D2, C, 0);
    cvt_bf16_k<<<dim3((unsigned)((long)B * N * D2 / 4 / 256)), dim3(256), 0, stream>>>(
        tgt_2dfeat, c.xA, (long)B * N * D2 / 4);
    gemm_bf16_wmma<<<dim3(C / 64, (unsigned)(B * N / 128)), blk, 0, stream>>>(c.xA, wdino, t2f, nullptr, D2, C, 0);

    // -------- 5 layers: [self, cross, self, cross, semantic] --------
    const int kind[5] = {0, 1, 0, 1, 2};
    for (int i = 0; i < 5; ++i) {
        if (kind[i] == 0) {          // self
            add_k<<<dim3((unsigned)((RC + 255) / 256)), dim3(256), 0, stream>>>(xs, s2,  RC);
            add_k<<<dim3((unsigned)((RC + 255) / 256)), dim3(256), 0, stream>>>(xt, t2f, RC);
            run_attn(c, xs, xs, xs, B, N, N, wqT[i], wkT[i], wvT[i], wmT[i], w1T[i], w2T[i]);
            run_attn(c, xt, xt, xt, B, N, N, wqT[i], wkT[i], wvT[i], wmT[i], w1T[i], w2T[i]);
        } else if (kind[i] == 1) {   // cross (tgt sees updated src)
            run_attn(c, xs, xt, xs, B, N, N, wqT[i], wkT[i], wvT[i], wmT[i], w1T[i], w2T[i]);
            run_attn(c, xt, xs, xt, B, N, N, wqT[i], wkT[i], wvT[i], wmT[i], w1T[i], w2T[i]);
        } else {                     // semantic subspace cross
            gather_k<<<dim3(G * Lp), dim3(256), 0, stream>>>(xs, si, sb,  N, Lp, G / B);
            gather_k<<<dim3(G * Lp), dim3(256), 0, stream>>>(xt, ti, tsb, N, Lp, G / B);
            run_attn(c, sb,  tsb, out_ss, G, Lp, Lp, wqT[i], wkT[i], wvT[i], wmT[i], w1T[i], w2T[i]);
            run_attn(c, tsb, sb,  out_ts, G, Lp, Lp, wqT[i], wkT[i], wvT[i], wmT[i], w1T[i], w2T[i]);
        }
    }

    hipMemcpyAsync(out_src, xs, RC * sizeof(float), hipMemcpyDeviceToDevice, stream);
    hipMemcpyAsync(out_tgt, xt, RC * sizeof(float), hipMemcpyDeviceToDevice, stream);
}
